// InterFrameAttention_5291399708690
// MI455X (gfx1250) — compile-verified, matrix operand order checked by reading, not run on hardware
//
#include <hip/hip_runtime.h>

// ---------------------------------------------------------------------------
// InterFrameAttention for MI455X (gfx1250, wave32, WMMA f16->f32)
// Shapes: B=4, N=2048, C=256, H=8, Dh=32, M=64, Mh=8
// GEMMs: LDS-staged tiles (coalesced b128 global loads, f16 conversion once,
// weights transposed in LDS so WMMA B-fragments are contiguous ds_load_b128).
// Attention: flash-style single pass, 5 WMMA per 32-key tile per wave;
// softmax row reductions via v_permlane16_b32 (VALU, bound_ctrl=1 so the
// 'old' operand is dead -> no v_mov materialization).
// ---------------------------------------------------------------------------

typedef _Float16 v16h __attribute__((ext_vector_type(16)));
typedef _Float16 v8h  __attribute__((ext_vector_type(8)));
typedef float    v8f  __attribute__((ext_vector_type(8)));

#define WMMA_F32_F16(A, Bm, Cm) \
  __builtin_amdgcn_wmma_f32_16x16x32_f16(false, (A), false, (Bm), (short)0, (Cm), false, false)

// A/B fragment element -> K mapping (16-bit 16x32 operand, wave32):
//   e in [0,8):  K = base + e          (base = 8 * (lane>>4))
//   e in [8,16): K = 16 + base + (e-8)
// -> per lane, two contiguous 8-half (16B) chunks at offsets base and 16+base.
__device__ __forceinline__ v16h frag_from_f16_row(const _Float16* __restrict__ row, int base) {
  v8h lo = *(const v8h*)(row + base);
  v8h hi = *(const v8h*)(row + 16 + base);
  v16h r;
#pragma unroll
  for (int i = 0; i < 8; ++i) { r[i] = lo[i]; r[i + 8] = hi[i]; }
  return r;
}

// --- VALU lane-xor within each 16-lane row via v_permlane16_b32 -------------
// select table: nibble j = j ^ mask (same table for both rows of 16).
// bound_ctrl=1: all selects are in-row and EXEC is all-ones, so OOB-fetch-0
// never triggers and the 'old' operand is dead (no mov needed).
__device__ __forceinline__ float permxor16(float x, unsigned s0, unsigned s1) {
  int v = __float_as_int(x);
  v = __builtin_amdgcn_permlane16(v, v, (int)s0, (int)s1, false, true);
  return __int_as_float(v);
}
__device__ __forceinline__ float rowmax16(float t) {
  t = fmaxf(t, permxor16(t, 0x67452301u, 0xEFCDAB89u));  // xor 1
  t = fmaxf(t, permxor16(t, 0x54761032u, 0xDCFE98BAu));  // xor 2
  t = fmaxf(t, permxor16(t, 0x32107654u, 0xBA98FEDCu));  // xor 4
  t = fmaxf(t, permxor16(t, 0xFEDCBA98u, 0x76543210u));  // xor 8
  return t;
}
__device__ __forceinline__ float rowsum16(float t) {
  t += permxor16(t, 0x67452301u, 0xEFCDAB89u);
  t += permxor16(t, 0x54761032u, 0xDCFE98BAu);
  t += permxor16(t, 0x32107654u, 0xBA98FEDCu);
  t += permxor16(t, 0xFEDCBA98u, 0x76543210u);
  return t;
}

// ---------------------------------------------------------------------------
// Kernel 1: Q/K/V projections, LDS-staged.  Z template: 0=Q, 1=K, 2=V.
// Block tile: 128 rows x 64 cols, K-step 32.  8 waves, each 16 rows x 64 cols.
// Outputs f16: Qh/Kh [bh, n, 32] (Q pre-scaled), Vth transposed [bh, 32, n].
// ---------------------------------------------------------------------------
template <int Z>
__global__ __launch_bounds__(256) void qkv_proj_kernel(
    const float* __restrict__ X, const float* __restrict__ W,
    _Float16* __restrict__ Out)
{
  __shared__ __align__(16) _Float16 Asub[128][32];  // 8 KB, [row][k]
  __shared__ __align__(16) _Float16 Bsub[64][32];   // 4 KB, [col][k] (transposed)

  const int tid  = threadIdx.x;
  const int wave = tid >> 5;
  const int lane = tid & 31;
  const int hi = lane >> 4, lx = lane & 15, base = hi << 3;

  const int row0    = blockIdx.x * 128;   // 8192 rows
  const int colbase = blockIdx.y * 64;    // 256 cols

  constexpr int ldb  = (Z == 0) ? 256 : 512;
  constexpr int coff = (Z == 2) ? 256 : 0;

  const v8f zv = {};
  v8f acc[4] = {zv, zv, zv, zv};

  for (int k0 = 0; k0 < 256; k0 += 32) {
    // stage A tile: 128 x 32 f32 -> f16 (coalesced float4 loads)
#pragma unroll
    for (int i = 0; i < 4; ++i) {
      const int idx = tid + 256 * i;              // 0..1023
      const int row = idx >> 3, f4 = idx & 7;
      const float4 v = *(const float4*)(X + (size_t)(row0 + row) * 256 + k0 + f4 * 4);
      Asub[row][f4 * 4 + 0] = (_Float16)v.x;
      Asub[row][f4 * 4 + 1] = (_Float16)v.y;
      Asub[row][f4 * 4 + 2] = (_Float16)v.z;
      Asub[row][f4 * 4 + 3] = (_Float16)v.w;
    }
    // stage B tile transposed: W[k0..k0+31][colbase..+63] -> Bsub[col][k]
#pragma unroll
    for (int i = 0; i < 2; ++i) {
      const int idx = tid + 256 * i;              // 0..511
      const int k = idx >> 4, f4 = idx & 15;
      const float4 v = *(const float4*)(W + (size_t)(k0 + k) * ldb + coff + colbase + f4 * 4);
      Bsub[f4 * 4 + 0][k] = (_Float16)v.x;
      Bsub[f4 * 4 + 1][k] = (_Float16)v.y;
      Bsub[f4 * 4 + 2][k] = (_Float16)v.z;
      Bsub[f4 * 4 + 3][k] = (_Float16)v.w;
    }
    __syncthreads();

    const v16h a = frag_from_f16_row(&Asub[(wave << 4) + lx][0], base);
#pragma unroll
    for (int c = 0; c < 4; ++c) {
      const v16h b = frag_from_f16_row(&Bsub[(c << 4) + lx][0], base);
      acc[c] = WMMA_F32_F16(a, b, acc[c]);
    }
    __syncthreads();
  }

  const float qscale = 0.17677669529663687f;  // 32^-0.5
#pragma unroll
  for (int c = 0; c < 4; ++c) {
#pragma unroll
    for (int r = 0; r < 8; ++r) {
      const int grow = row0 + (wave << 4) + r + base;
      const int gcol = colbase + (c << 4) + lx;
      const int b_ = grow >> 11, n_ = grow & 2047;
      const int h_ = gcol >> 5,  d_ = gcol & 31;
      const float v = acc[c][r];
      if (Z == 0)
        Out[(((size_t)(b_ * 8 + h_) * 2048 + n_) << 5) + d_] = (_Float16)(v * qscale);
      else if (Z == 1)
        Out[(((size_t)(b_ * 8 + h_) * 2048 + n_) << 5) + d_] = (_Float16)v;
      else
        Out[(((size_t)(b_ * 8 + h_) * 32 + d_) << 11) + n_] = (_Float16)v;
    }
  }
}

// ---------------------------------------------------------------------------
// Kernel 2: cor_embed = cor @ Wcor  (K=3, tiny).  Writes f32 ce [B*N, 64] and
// f16 transposed CEth [bh, 16, n] with rows 8..15 zero-padded.
// ---------------------------------------------------------------------------
__global__ void cor_embed_kernel(const float* __restrict__ cor, const float* __restrict__ Wcor,
                                 float* __restrict__ ce, _Float16* __restrict__ CEth)
{
  const int idx = blockIdx.x * blockDim.x + threadIdx.x;  // 8192 * 128
  if (idx >= 8192 * 128) return;
  const int bn = idx >> 7, m2 = idx & 127;
  const int h = m2 >> 4, d = m2 & 15;
  const int b = bn >> 11, n = bn & 2047;
  const size_t tpos = (((size_t)(b * 8 + h) * 16 + d) << 11) + n;
  if (d < 8) {
    const int m = h * 8 + d;
    const float* c3 = cor + (size_t)bn * 3;
    const float v = c3[0] * Wcor[m] + c3[1] * Wcor[64 + m] + c3[2] * Wcor[128 + m];
    ce[(size_t)bn * 64 + m] = v;
    CEth[tpos] = (_Float16)v;
  } else {
    CEth[tpos] = (_Float16)0.f;
  }
}

// ---------------------------------------------------------------------------
// Kernel 3: flash attention.  One 16-query tile per wave, 8 waves/WG,
// grid (N/128, B*H).  Online softmax; 5 WMMA per 32-key tile.
// ---------------------------------------------------------------------------
__global__ __launch_bounds__(256) void attn_kernel(
    const _Float16* __restrict__ Qh, const _Float16* __restrict__ Kh,
    const _Float16* __restrict__ Vth, const _Float16* __restrict__ CEth,
    float* __restrict__ attn_x, float* __restrict__ c_rev)
{
  __shared__ __align__(16) _Float16 plds[8][16][40];  // per-wave P tile, padded rows

  const int wave = threadIdx.x >> 5;
  const int lane = threadIdx.x & 31;
  const int hi = lane >> 4, lx = lane & 15, base = hi << 3;
  const int bh = blockIdx.y;
  const int q0 = blockIdx.x * 128 + wave * 16;

  const _Float16* Qb = Qh   + (size_t)bh * (2048 * 32);
  const _Float16* Kb = Kh   + (size_t)bh * (2048 * 32);
  const _Float16* Vb = Vth  + (size_t)bh * (32 * 2048);
  const _Float16* Cb = CEth + (size_t)bh * (16 * 2048);

  const v16h qf = frag_from_f16_row(Qb + (size_t)(q0 + lx) * 32, base);

  v8f o0 = {}, o1 = {}, crv = {};
  float rm[8], rl[8];
#pragma unroll
  for (int r = 0; r < 8; ++r) { rm[r] = -3.0e38f; rl[r] = 0.f; }

  for (int j0 = 0; j0 < 2048; j0 += 32) {
    // S = Q * K^T  (scale folded into Q)
    const v16h kb0 = frag_from_f16_row(Kb + (size_t)(j0 + lx) * 32, base);
    const v16h kb1 = frag_from_f16_row(Kb + (size_t)(j0 + 16 + lx) * 32, base);
    v8f s0 = {}, s1 = {};
    s0 = WMMA_F32_F16(qf, kb0, s0);
    s1 = WMMA_F32_F16(qf, kb1, s1);

    // online softmax per row (row r+8*hi lives on this half-wave's lanes)
#pragma unroll
    for (int r = 0; r < 8; ++r) {
      const float t  = rowmax16(fmaxf(s0[r], s1[r]));
      const float mn = fmaxf(rm[r], t);
      const float alpha = __expf(rm[r] - mn);
      const float p0 = __expf(s0[r] - mn);
      const float p1 = __expf(s1[r] - mn);
      const float rs = rowsum16(p0 + p1);
      rl[r] = rl[r] * alpha + rs;
      rm[r] = mn;
      o0[r] *= alpha; o1[r] *= alpha; crv[r] *= alpha;
      plds[wave][r + base][lx]      = (_Float16)p0;  // C-layout -> LDS
      plds[wave][r + base][lx + 16] = (_Float16)p1;
    }

    // P tile back as A fragment (same wave; compiler orders via s_wait_dscnt)
    const v16h pf  = frag_from_f16_row(&plds[wave][lx][0], base);
    const v16h vb0 = frag_from_f16_row(Vb + ((size_t)lx << 11) + j0, base);
    const v16h vb1 = frag_from_f16_row(Vb + ((size_t)(lx + 16) << 11) + j0, base);
    const v16h cb  = frag_from_f16_row(Cb + ((size_t)lx << 11) + j0, base);
    o0  = WMMA_F32_F16(pf, vb0, o0);
    o1  = WMMA_F32_F16(pf, vb1, o1);
    crv = WMMA_F32_F16(pf, cb, crv);
  }

  const int b_ = bh >> 3, h_ = bh & 7;
#pragma unroll
  for (int r = 0; r < 8; ++r) {
    const int   q   = q0 + r + base;
    const float inv = 1.0f / rl[r];
    float* xr = attn_x + (((size_t)b_ * 2048 + q) << 8) + (h_ << 5);
    xr[lx]      = o0[r] * inv;
    xr[lx + 16] = o1[r] * inv;
    if (lx < 8)
      c_rev[(((size_t)b_ * 2048 + q) << 6) + (h_ << 3) + lx] = crv[r] * inv;
  }
}

// ---------------------------------------------------------------------------
// Kernel 4: x_out = attn_x @ Wproj + bproj   [8192,256]x[256,256], LDS-staged.
// ---------------------------------------------------------------------------
__global__ __launch_bounds__(256) void proj_kernel(
    const float* __restrict__ attn_x, const float* __restrict__ Wproj,
    const float* __restrict__ bproj, float* __restrict__ out)
{
  __shared__ __align__(16) _Float16 Asub[128][32];
  __shared__ __align__(16) _Float16 Bsub[64][32];

  const int tid  = threadIdx.x;
  const int wave = tid >> 5;
  const int lane = tid & 31;
  const int hi = lane >> 4, lx = lane & 15, base = hi << 3;
  const int row0 = blockIdx.x * 128;
  const int colbase = blockIdx.y * 64;

  const v8f zv = {};
  v8f acc[4] = {zv, zv, zv, zv};

  for (int k0 = 0; k0 < 256; k0 += 32) {
#pragma unroll
    for (int i = 0; i < 4; ++i) {
      const int idx = tid + 256 * i;
      const int row = idx >> 3, f4 = idx & 7;
      const float4 v = *(const float4*)(attn_x + (size_t)(row0 + row) * 256 + k0 + f4 * 4);
      Asub[row][f4 * 4 + 0] = (_Float16)v.x;
      Asub[row][f4 * 4 + 1] = (_Float16)v.y;
      Asub[row][f4 * 4 + 2] = (_Float16)v.z;
      Asub[row][f4 * 4 + 3] = (_Float16)v.w;
    }
#pragma unroll
    for (int i = 0; i < 2; ++i) {
      const int idx = tid + 256 * i;
      const int k = idx >> 4, f4 = idx & 15;
      const float4 v = *(const float4*)(Wproj + (size_t)(k0 + k) * 256 + colbase + f4 * 4);
      Bsub[f4 * 4 + 0][k] = (_Float16)v.x;
      Bsub[f4 * 4 + 1][k] = (_Float16)v.y;
      Bsub[f4 * 4 + 2][k] = (_Float16)v.z;
      Bsub[f4 * 4 + 3][k] = (_Float16)v.w;
    }
    __syncthreads();

    const v16h a = frag_from_f16_row(&Asub[(wave << 4) + lx][0], base);
#pragma unroll
    for (int c = 0; c < 4; ++c) {
      const v16h b = frag_from_f16_row(&Bsub[(c << 4) + lx][0], base);
      acc[c] = WMMA_F32_F16(a, b, acc[c]);
    }
    __syncthreads();
  }

#pragma unroll
  for (int c = 0; c < 4; ++c) {
#pragma unroll
    for (int r = 0; r < 8; ++r) {
      const int grow = row0 + (wave << 4) + r + base;
      const int gcol = colbase + (c << 4) + lx;
      out[((size_t)grow << 8) + gcol] = acc[c][r] + bproj[gcol];
    }
  }
}

// ---------------------------------------------------------------------------
// Kernel 5: motion = (c_rev - ce) @ Wm + bm   [8192,64]x[64,64].
// Wm^T staged once in LDS; each wave: 16 rows x 64 cols, K=64 in 2 steps.
// ---------------------------------------------------------------------------
__global__ __launch_bounds__(256) void motion_kernel(
    const float* __restrict__ c_rev, const float* __restrict__ ce,
    const float* __restrict__ Wm, const float* __restrict__ bm, float* __restrict__ out)
{
  __shared__ __align__(16) _Float16 Wt[64][64];  // 8 KB, [col][k]

  const int tid  = threadIdx.x;
  const int wave = tid >> 5;
  const int lane = tid & 31;
  const int hi = lane >> 4, lx = lane & 15, base = hi << 3;
  const int row0 = blockIdx.x * 128;

#pragma unroll
  for (int i = 0; i < 4; ++i) {
    const int idx = tid + 256 * i;              // 0..1023
    const int k = idx >> 4, f4 = idx & 15;
    const float4 v = *(const float4*)(Wm + (size_t)k * 64 + f4 * 4);
    Wt[f4 * 4 + 0][k] = (_Float16)v.x;
    Wt[f4 * 4 + 1][k] = (_Float16)v.y;
    Wt[f4 * 4 + 2][k] = (_Float16)v.z;
    Wt[f4 * 4 + 3][k] = (_Float16)v.w;
  }
  __syncthreads();

  const float* ar = c_rev + (size_t)(row0 + (wave << 4) + lx) * 64;
  const float* er = ce    + (size_t)(row0 + (wave << 4) + lx) * 64;

  const v8f zv = {};
  v8f acc[4] = {zv, zv, zv, zv};
#pragma unroll
  for (int k0 = 0; k0 < 64; k0 += 32) {
    v16h a;
#pragma unroll
    for (int i = 0; i < 8; ++i) {
      const int ka = k0 + base + i, kb = k0 + 16 + base + i;
      a[i]     = (_Float16)(ar[ka] - er[ka]);
      a[i + 8] = (_Float16)(ar[kb] - er[kb]);
    }
#pragma unroll
    for (int c = 0; c < 4; ++c) {
      const v16h b = frag_from_f16_row(&Wt[(c << 4) + lx][k0], base);
      acc[c] = WMMA_F32_F16(a, b, acc[c]);
    }
  }

#pragma unroll
  for (int c = 0; c < 4; ++c) {
#pragma unroll
    for (int r = 0; r < 8; ++r) {
      const int grow = row0 + (wave << 4) + r + base;
      const int gcol = (c << 4) + lx;
      out[((size_t)grow << 6) + gcol] = acc[c][r] + bm[gcol];
    }
  }
}

// ---------------------------------------------------------------------------
extern "C" void kernel_launch(void* const* d_in, const int* in_sizes, int n_in,
                              void* d_out, int out_size, void* d_ws, size_t ws_size,
                              hipStream_t stream) {
  (void)in_sizes; (void)n_in; (void)out_size; (void)ws_size;
  const float* x1    = (const float*)d_in[0];
  const float* x2    = (const float*)d_in[1];
  const float* cor   = (const float*)d_in[2];
  const float* Wq    = (const float*)d_in[3];
  const float* Wkv   = (const float*)d_in[4];
  const float* Wcor  = (const float*)d_in[5];
  const float* Wproj = (const float*)d_in[6];
  const float* bproj = (const float*)d_in[7];
  const float* Wm    = (const float*)d_in[8];
  const float* bm    = (const float*)d_in[9];

  char* ws = (char*)d_ws;                                  // 26 MB total
  _Float16* Qh     = (_Float16*)(ws);                      //  4 MB
  _Float16* Kh     = (_Float16*)(ws + ( 4u << 20));        //  4 MB
  _Float16* Vth    = (_Float16*)(ws + ( 8u << 20));        //  4 MB
  _Float16* CEth   = (_Float16*)(ws + (12u << 20));        //  2 MB
  float*    ce     = (float*)   (ws + (14u << 20));        //  2 MB
  float*    attn_x = (float*)   (ws + (16u << 20));        //  8 MB
  float*    c_rev  = (float*)   (ws + (24u << 20));        //  2 MB

  float* out_x      = (float*)d_out;                       // [4,2048,256]
  float* out_motion = out_x + (size_t)4 * 2048 * 256;      // [4,2048,64]

  qkv_proj_kernel<0><<<dim3(64, 4), 256, 0, stream>>>(x1, Wq,  Qh);
  qkv_proj_kernel<1><<<dim3(64, 4), 256, 0, stream>>>(x2, Wkv, Kh);
  qkv_proj_kernel<2><<<dim3(64, 4), 256, 0, stream>>>(x2, Wkv, Vth);
  cor_embed_kernel<<<(8192 * 128) / 256, 256, 0, stream>>>(cor, Wcor, ce, CEth);
  attn_kernel     <<<dim3(16, 32),    256, 0, stream>>>(Qh, Kh, Vth, CEth, attn_x, c_rev);
  proj_kernel     <<<dim3(64, 4),     256, 0, stream>>>(attn_x, Wproj, bproj, out_x);
  motion_kernel   <<<dim3(64, 1),     256, 0, stream>>>(c_rev, ce, Wm, bm, out_motion);
}
